// QKVProjectionLayer_51977694216468
// MI455X (gfx1250) — matrix operand order repaired and checked by reading, not compiled, not run on hardware
//
#include <hip/hip_runtime.h>
#include <stdint.h>

// ---------------------------------------------------------------------------
// QKV projection: out = hidden[16384x2048] @ w_qkv[2048x6144] + bias,
// scattered into q/k/v (B, heads, T, head_dim). Compute-bound -> bf16 WMMA.
// Pass 1: convert A->bf16; convert+transpose W->bf16 [N][K]   (into d_ws)
// Pass 2: GEMM, double-buffered LDS fed by GLOBAL_LOAD_ASYNC_TO_LDS_B128.
// ---------------------------------------------------------------------------

typedef __attribute__((ext_vector_type(16))) __bf16 v16bf;
typedef __attribute__((ext_vector_type(8)))  float  v8f;

#define GEMM_M 16384
#define GEMM_K 2048
#define GEMM_N 6144

// ---------------- helpers ----------------
__device__ __forceinline__ uint16_t f2bf_u16(float f) {
  union { __bf16 h; uint16_t u; } c;
  c.h = (__bf16)f;              // hardware RNE convert
  return c.u;
}
__device__ __forceinline__ uint32_t pack_bf16(float a, float b) {
  return (uint32_t)f2bf_u16(a) | ((uint32_t)f2bf_u16(b) << 16);
}

// LDS byte address = low 32 bits of flat shared pointer (LDS aperture rule)
__device__ __forceinline__ uint32_t lds_addr_of(const void* p) {
  return (uint32_t)(uintptr_t)p;
}

// async memory -> LDS copy, 16 bytes per lane, tracked by ASYNCcnt
__device__ __forceinline__ void async_b128(uint32_t lds, const void* gaddr) {
  asm volatile("global_load_async_to_lds_b128 %0, %1, off"
               :: "v"(lds), "v"(gaddr) : "memory");
}
__device__ __forceinline__ void wait_async_le4() {
  asm volatile("s_wait_asynccnt 0x4" ::: "memory");
}
__device__ __forceinline__ void wait_async_0() {
  asm volatile("s_wait_asynccnt 0x0" ::: "memory");
}

// ---------------- pass 1a: hidden fp32 -> bf16 (same layout) ----------------
__global__ __launch_bounds__(256)
void convert_a_bf16(const float* __restrict__ A, uint16_t* __restrict__ Abf) {
  size_t idx  = (size_t)blockIdx.x * 256 + threadIdx.x;   // 8 elems per thread
  size_t base = idx * 8;
  float4 v0 = *(const float4*)(A + base);
  float4 v1 = *(const float4*)(A + base + 4);
  uint4 p;
  p.x = pack_bf16(v0.x, v0.y);
  p.y = pack_bf16(v0.z, v0.w);
  p.z = pack_bf16(v1.x, v1.y);
  p.w = pack_bf16(v1.z, v1.w);
  *(uint4*)(Abf + base) = p;
}

// ------------- pass 1b: W [K][N] fp32 -> Wt [N][K] bf16 (64x64 tiles) -------
__global__ __launch_bounds__(256)
void transpose_w_bf16(const float* __restrict__ W, uint16_t* __restrict__ Wt) {
  __shared__ __align__(16) uint16_t tile[64][72];
  const int t  = threadIdx.x;
  const int n0 = blockIdx.x * 64;
  const int k0 = blockIdx.y * 64;
#pragma unroll
  for (int i = 0; i < 4; ++i) {
    int idx = t + i * 256;
    int r   = idx >> 4;
    int c   = (idx & 15) * 4;
    float4 v = *(const float4*)(W + (size_t)(k0 + r) * GEMM_N + n0 + c);
    tile[r][c + 0] = f2bf_u16(v.x);
    tile[r][c + 1] = f2bf_u16(v.y);
    tile[r][c + 2] = f2bf_u16(v.z);
    tile[r][c + 3] = f2bf_u16(v.w);
  }
  __syncthreads();
#pragma unroll
  for (int i = 0; i < 4; ++i) {
    int idx = t + i * 256;
    int rn  = idx >> 4;
    int ck  = (idx & 15) * 4;
    uint2 p;
    p.x = (uint32_t)tile[ck + 0][rn] | ((uint32_t)tile[ck + 1][rn] << 16);
    p.y = (uint32_t)tile[ck + 2][rn] | ((uint32_t)tile[ck + 3][rn] << 16);
    *(uint2*)(Wt + (size_t)(n0 + rn) * GEMM_K + k0 + ck) = p;
  }
}

// ---------------- pass 2: bf16 GEMM, async double-buffered LDS --------------
#define TM 128
#define TN 128
#define TK 32
#define LDSTR 40   // halfwords per LDS row (32 data + 8 pad); 80 B, 16B-aligned

__global__ __launch_bounds__(256)
void qkv_gemm_bf16(const uint16_t* __restrict__ Abf,  // [16384][2048] bf16
                   const uint16_t* __restrict__ Wt,   // [6144][2048] bf16 (N-major)
                   const float* __restrict__ bias,
                   float* __restrict__ out)
{
  __shared__ __align__(16) uint16_t As[2][TM * LDSTR];  // 2 x 10 KB
  __shared__ __align__(16) uint16_t Bs[2][TN * LDSTR];  // 2 x 10 KB

  const int t    = threadIdx.x;
  const int lane = t & 31;
  const int wave = t >> 5;
  const int wm   = wave & 3;    // 4 waves along M (32 rows each)
  const int wn   = wave >> 2;   // 2 waves along N (64 cols each)
  const int m0   = blockIdx.y * TM;
  const int n0   = blockIdx.x * TN;
  const int mlo  = lane & 15;
  const int hi   = lane >> 4;

  // staging: 512 b128 chunks per tile (128 rows x 4 segs); 2 chunks per thread
  const int row0 = t >> 2, seg = t & 3;
  const int row1 = row0 + 64;               // (t+256)>>2

  const uint16_t* ga0 = Abf + (size_t)(m0 + row0) * GEMM_K + seg * 8;
  const uint16_t* ga1 = Abf + (size_t)(m0 + row1) * GEMM_K + seg * 8;
  const uint16_t* gb0 = Wt  + (size_t)(n0 + row0) * GEMM_K + seg * 8;
  const uint16_t* gb1 = Wt  + (size_t)(n0 + row1) * GEMM_K + seg * 8;

  const uint32_t la0 = lds_addr_of(&As[0][row0 * LDSTR + seg * 8]);
  const uint32_t la1 = lds_addr_of(&As[0][row1 * LDSTR + seg * 8]);
  const uint32_t lb0 = lds_addr_of(&Bs[0][row0 * LDSTR + seg * 8]);
  const uint32_t lb1 = lds_addr_of(&Bs[0][row1 * LDSTR + seg * 8]);
  const uint32_t bufstep = (uint32_t)(TM * LDSTR * 2);  // 10240 B

  // per-lane fragment offsets (halfword indices into an LDS buffer)
  int aoff[2], boff[4];
#pragma unroll
  for (int am = 0; am < 2; ++am)
    aoff[am] = (wm * 32 + am * 16 + mlo) * LDSTR + hi * 8;
#pragma unroll
  for (int bn = 0; bn < 4; ++bn)
    boff[bn] = (wn * 64 + bn * 16 + mlo) * LDSTR + hi * 8;

  v8f acc[2][4];
  v8f vzero = {0.f, 0.f, 0.f, 0.f, 0.f, 0.f, 0.f, 0.f};
#pragma unroll
  for (int i = 0; i < 2; ++i)
#pragma unroll
    for (int j = 0; j < 4; ++j) acc[i][j] = vzero;

  // prologue: async-stage tile 0 into buffer 0
  async_b128(la0, ga0);
  async_b128(la1, ga1);
  async_b128(lb0, gb0);
  async_b128(lb1, gb1);

  const int NITER = GEMM_K / TK;   // 64
  for (int kt = 0; kt < NITER; ++kt) {
    const int cur = kt & 1;
    if (kt + 1 < NITER) {
      // async-stage next tile into the other buffer
      const size_t kadv = (size_t)(kt + 1) * TK;            // halfword advance
      const uint32_t lb  = (uint32_t)((kt + 1) & 1) * bufstep;
      async_b128(la0 + lb, ga0 + kadv);
      async_b128(la1 + lb, ga1 + kadv);
      async_b128(lb0 + lb, gb0 + kadv);
      async_b128(lb1 + lb, gb1 + kadv);
      wait_async_le4();     // tile kt's 4 ops complete (in-order)
    } else {
      wait_async_0();
    }
    __syncthreads();        // whole tile visible to all waves

    union Frag { v16bf v; uint4 u[2]; };
    Frag a[2], b[4];
#pragma unroll
    for (int am = 0; am < 2; ++am) {
      // A 16x32 bf16 layout: lanes 0-15 M=lane, K 0-7/16-23; lanes 16-31 K 8-15/24-31
      a[am].u[0] = *(const uint4*)&As[cur][aoff[am]];
      a[am].u[1] = *(const uint4*)&As[cur][aoff[am] + 16];
    }
#pragma unroll
    for (int bn = 0; bn < 4; ++bn) {
      // B 32x16 bf16 layout: lanes 0-15 K 0-7/16-23, lanes 16-31 K 8-15/24-31, N=lane&15
      b[bn].u[0] = *(const uint4*)&Bs[cur][boff[bn]];
      b[bn].u[1] = *(const uint4*)&Bs[cur][boff[bn] + 16];
    }
#pragma unroll
    for (int am = 0; am < 2; ++am)
#pragma unroll
      for (int bn = 0; bn < 4; ++bn)
        acc[am][bn] = __builtin_amdgcn_wmma_f32_16x16x32_bf16(
            false, a[am].v, false, b[bn].v, (short)0, acc[am][bn], false, false);

    __syncthreads();        // readers done before next iter's async overwrites
  }

  // ---- epilogue: + bias, scatter into (which, b, head, t, hd) layout ----
  const int64_t plane = (int64_t)4 * 16 * 4096 * 128;
#pragma unroll
  for (int am = 0; am < 2; ++am) {
#pragma unroll
    for (int bn = 0; bn < 4; ++bn) {
      int n = n0 + wn * 64 + bn * 16 + mlo;
      float bv = bias[n];
      int which = n >> 11;
      int rem   = n & 2047;
      int head  = rem >> 7;
      int hd    = rem & 127;
#pragma unroll
      for (int r = 0; r < 8; ++r) {
        int m  = m0 + wm * 32 + am * 16 + r + (hi << 3);
        int bb = m >> 12;
        int tt = m & 4095;
        int64_t oi = (int64_t)which * plane +
                     (((int64_t)(bb * 16 + head)) * 4096 + tt) * 128 + hd;
        out[oi] = acc[am][bn][r] + bv;
      }
    }
  }
}

// ---------------- fallback: fused convert-in-loop GEMM ----------------------
#define FLDA 40
#define FLDB 40
__global__ __launch_bounds__(256)
void qkv_wmma_fused(const float* __restrict__ A, const float* __restrict__ W,
                    const float* __restrict__ bias, float* __restrict__ out)
{
  __shared__ __align__(16) uint16_t Asf[128 * FLDA];
  __shared__ __align__(16) uint16_t Bsf[128 * FLDB];
  const int t = threadIdx.x, lane = t & 31, wave = t >> 5;
  const int wm = wave & 3, wn = wave >> 2;
  const int m0 = blockIdx.y * 128, n0 = blockIdx.x * 128;
  const int mlo = lane & 15, hi = lane >> 4;
  v8f acc[2][4];
  v8f vzero = {0.f,0.f,0.f,0.f,0.f,0.f,0.f,0.f};
#pragma unroll
  for (int i = 0; i < 2; ++i)
#pragma unroll
    for (int j = 0; j < 4; ++j) acc[i][j] = vzero;
  for (int k0 = 0; k0 < GEMM_K; k0 += 32) {
#pragma unroll
    for (int i = 0; i < 4; ++i) {
      int idx = t + i * 256, row = idx >> 3, seg = idx & 7;
      float4 v = *(const float4*)(A + (size_t)(m0 + row) * GEMM_K + k0 + seg * 4);
      uint32_t* dst = (uint32_t*)&Asf[row * FLDA + seg * 4];
      dst[0] = pack_bf16(v.x, v.y);
      dst[1] = pack_bf16(v.z, v.w);
    }
#pragma unroll
    for (int i = 0; i < 4; ++i) {
      int idx = t + i * 256, kk = idx >> 5, nn = (idx & 31) * 4;
      float4 v = *(const float4*)(W + (size_t)(k0 + kk) * GEMM_N + n0 + nn);
      Bsf[(nn + 0) * FLDB + kk] = f2bf_u16(v.x);
      Bsf[(nn + 1) * FLDB + kk] = f2bf_u16(v.y);
      Bsf[(nn + 2) * FLDB + kk] = f2bf_u16(v.z);
      Bsf[(nn + 3) * FLDB + kk] = f2bf_u16(v.w);
    }
    __syncthreads();
    union Frag { v16bf v; uint4 u[2]; };
    Frag a[2], b[4];
#pragma unroll
    for (int am = 0; am < 2; ++am) {
      int m = wm * 32 + am * 16 + mlo, kb = hi * 8;
      a[am].u[0] = *(const uint4*)&Asf[m * FLDA + kb];
      a[am].u[1] = *(const uint4*)&Asf[m * FLDA + kb + 16];
    }
#pragma unroll
    for (int bn = 0; bn < 4; ++bn) {
      int n = wn * 64 + bn * 16 + mlo, kb = hi * 8;
      b[bn].u[0] = *(const uint4*)&Bsf[n * FLDB + kb];
      b[bn].u[1] = *(const uint4*)&Bsf[n * FLDB + kb + 16];
    }
#pragma unroll
    for (int am = 0; am < 2; ++am)
#pragma unroll
      for (int bn = 0; bn < 4; ++bn)
        acc[am][bn] = __builtin_amdgcn_wmma_f32_16x16x32_bf16(
            false, a[am].v, false, b[bn].v, (short)0, acc[am][bn], false, false);
    __syncthreads();
  }
  const int64_t plane = (int64_t)4 * 16 * 4096 * 128;
#pragma unroll
  for (int am = 0; am < 2; ++am)
#pragma unroll
    for (int bn = 0; bn < 4; ++bn) {
      int n = n0 + wn * 64 + bn * 16 + mlo;
      float bv = bias[n];
      int which = n >> 11, rem = n & 2047, head = rem >> 7, hd = rem & 127;
#pragma unroll
      for (int r = 0; r < 8; ++r) {
        int m = m0 + wm * 32 + am * 16 + r + (hi << 3);
        int bb = m >> 12, tt = m & 4095;
        int64_t oi = (int64_t)which * plane +
                     (((int64_t)(bb * 16 + head)) * 4096 + tt) * 128 + hd;
        out[oi] = acc[am][bn][r] + bv;
      }
    }
}

// ---------------- host launcher ----------------
extern "C" void kernel_launch(void* const* d_in, const int* in_sizes, int n_in,
                              void* d_out, int out_size, void* d_ws, size_t ws_size,
                              hipStream_t stream) {
  const float* hidden = (const float*)d_in[0];
  const float* w_qkv  = (const float*)d_in[1];
  const float* b_qkv  = (const float*)d_in[2];
  float* out          = (float*)d_out;

  const size_t needA = (size_t)GEMM_M * GEMM_K * sizeof(uint16_t);  // 64 MB
  const size_t needW = (size_t)GEMM_K * GEMM_N * sizeof(uint16_t);  // 24 MB

  if (ws_size >= needA + needW) {
    uint16_t* Abf = (uint16_t*)d_ws;
    uint16_t* Wt  = (uint16_t*)((char*)d_ws + needA);
    convert_a_bf16<<<(GEMM_M * (size_t)GEMM_K) / (256 * 8), 256, 0, stream>>>(hidden, Abf);
    transpose_w_bf16<<<dim3(GEMM_N / 64, GEMM_K / 64), 256, 0, stream>>>(w_qkv, Wt);
    qkv_gemm_bf16<<<dim3(GEMM_N / TN, GEMM_M / TM), 256, 0, stream>>>(Abf, Wt, b_qkv, out);
  } else {
    qkv_wmma_fused<<<dim3(GEMM_N / 128, GEMM_M / 128), 256, 0, stream>>>(hidden, w_qkv, b_qkv, out);
  }
}
